// VolumeRenderer_51908974739815
// MI455X (gfx1250) — matrix operand order repaired and checked by reading, not compiled, not run on hardware
//
#include <hip/hip_runtime.h>

// ---------------------------------------------------------------------------
// NeRF coarse+fine volume renderer for MI455X (gfx1250, wave32).
//
// Roofline: ~412 MB HBM traffic, ~0.5 GFLOP -> pure bandwidth problem
// (floor ~18us @ 23.3 TB/s). One wave32 per 16 rays. Sigma/rgb are streamed
// once as 2D tiles into LDS (per-ray access is a transpose). On toolchains
// that expose it, the Tensor Data Mover (tensor_load_to_lds) does the staging
// asynchronously with hardware LDS padding and double buffering, synchronized
// with s_wait_tensorcnt. The exclusive log-transmittance scan
// (cumprod(x) = exp(cumsum(log x))) runs as a real 16x16 matmul on
// V_WMMA_F32_16X16X32_F16 against constant strictly-lower-triangular B.
// ---------------------------------------------------------------------------

typedef __attribute__((ext_vector_type(16))) _Float16 v16h;
typedef __attribute__((ext_vector_type(8)))  float    v8f;
typedef __attribute__((ext_vector_type(4)))  unsigned int u32x4;
typedef __attribute__((ext_vector_type(8)))  int      i32x8;
typedef __attribute__((ext_vector_type(4)))  int      i32x4;

#define NCOARSE 64
#define NTOTAL  128

// LDS pitches (64 banks, float4 aligned). Sigma rows: 32 payload + 4 pad = 36.
// Rgb rows: 96 payload + 3x4 interleaved pad (one per 32 dwords) = 108.
#define P_SIG 36
#define P_RGB 108
#define P_W   68
#define P_Z   133

// rgb LDS address map: payload offset j -> j + 4*(j/32)  (pad every 32 dwords)
__device__ __forceinline__ int rgb_idx(int r, int j) {
  return r * P_RGB + j + ((j >> 5) << 2);
}

struct SharedBufs {
  float work[2][16 * P_SIG];   // sigma chunk (double buffered); overwritten with e=exp(-d*relu(s))
  float rgb [2][16 * P_RGB];   // rgb chunk (double buffered); reused as CDF between passes
  float w   [16 * P_W];        // coarse weights; reused as z_fine
  float z   [16 * P_Z];        // merged sorted z (fine pass)
  float nearL[16], dL[16], farL[16], SL[16], invSL[16];
};

// ---------------------------------------------------------------------------
// Tensor Data Mover: async 2D tile (global -> LDS) with hardware padding.
// D# packing per CDNA5 ISA 8.3/8.4: group0 = {count=1, lds_addr,
// global_addr[56:0], type=2}; group1 = {data_size=4B, pad_enable,
// pad_interval=32dw, pad_amount=4dw, tensor_dim0/1 = tile dims,
// tile_dim0/1, tensor_dim0_stride}. Groups 2/3 zero (2D tensor).
// ---------------------------------------------------------------------------
#if __has_builtin(__builtin_amdgcn_tensor_load_to_lds)
#define USE_TDM 1
__device__ __forceinline__ void tdm_load_2d(const void* gaddr, unsigned ldsOff,
                                            unsigned tileD0, unsigned tileD1,
                                            unsigned rowStrideElems) {
  unsigned long long ga = (unsigned long long)gaddr;
  u32x4 g0 = {};
  g0[0] = 1u;                                             // count=1 (user D#)
  g0[1] = ldsOff;                                         // lds_addr (bytes)
  g0[2] = (unsigned)(ga & 0xFFFFFFFFull);                 // global_addr[31:0]
  g0[3] = (unsigned)((ga >> 32) & 0x1FFFFFFull)           // global_addr[56:32]
        | (2u << 30);                                     // type = 2 ("image")
  i32x8 g1 = {};
  g1[0] = (int)((2u << 16)                                // data_size = 4B
              | (1u << 20)                                // pad_enable
              | (4u << 22)                                // pad_interval: 32 dwords
              | (3u << 25));                              // pad_amount: 4 dwords
  g1[1] = (int)((tileD0 & 0xFFFFu) << 16);                // tensor_dim0[15:0]
  g1[2] = (int)((tileD1 & 0xFFFFu) << 16);                // tensor_dim1[15:0]
  g1[3] = (int)((tileD0 & 0xFFFFu) << 16);                // tile_dim0
  g1[4] = (int)(tileD1 & 0xFFFFu);                        // tile_dim1 (tile_dim2=0)
  g1[5] = (int)rowStrideElems;                            // tensor_dim0_stride[31:0]
  g1[6] = 0; g1[7] = 0;
  i32x4 z4 = {};
#if __clang_major__ >= 23
  i32x8 z8 = {};
  __builtin_amdgcn_tensor_load_to_lds(g0, g1, z4, z4, z8, 0);
#else
  __builtin_amdgcn_tensor_load_to_lds(g0, g1, z4, z4, 0);
#endif
}
__device__ __forceinline__ unsigned lds_addr_of(const void* p) {
  // LDS generic pointers carry the allocation-relative offset in addr[31:0].
  return (unsigned)(unsigned long long)p;
}
#else
#define USE_TDM 0
#endif

// ---------------------------------------------------------------------------
// One composite pass over KTOT samples for 16 rays (one wave).
// Layouts (CDNA5 ISA 7.12.2), L = lane, r = L&15, h = L>>4:
//   A (f16 16x32):  element e of lane (r,h) holds sample k = e+8h (+8 if e>=8)
//   B (f16 32x16):  element i of lane holds row t = i+8h (+8 if i>=8), column n = L&15
//   C/D (f32 16x16): element v of lane holds row M = v+8h, column N = L&15
// ---------------------------------------------------------------------------
template <int KTOT, bool COARSE>
__device__ __forceinline__ void composite_pass(
    int L,
    const float* __restrict__ gSig,   // wave-base sigma  [16][KTOT]
    const float* __restrict__ gRgb,   // wave-base rgb    [16][KTOT][3]
    SharedBufs& S, v16h B0, v16h B1,
    float* __restrict__ out, int ray0, int nRays)
{
  const int r = L & 15;
  const int h = L >> 4;

  float acc_w[8], acc_r[8], acc_g[8], acc_b[8], acc_z[8];
#pragma unroll
  for (int v = 0; v < 8; ++v) { acc_w[v] = acc_r[v] = acc_g[v] = acc_b[v] = acc_z[v] = 0.0f; }

  float carry = 0.0f;              // lane L holds log-carry for ray (L&15)
  constexpr int NCHUNK = KTOT / 32;

#if USE_TDM
  // Pre-issue chunk 0 (2 tensor ops: sigma tile + rgb tile).
  tdm_load_2d(gSig, lds_addr_of(&S.work[0][0]), 32, 16, KTOT);
  tdm_load_2d(gRgb, lds_addr_of(&S.rgb[0][0]),  96, 16, KTOT * 3);
#endif

  for (int c = 0; c < NCHUNK; ++c) {
    float* WK = &S.work[c & 1][0];
    float* RB = &S.rgb [c & 1][0];

#if USE_TDM
    // Pipeline: issue chunk c+1 into the other buffer, then wait for chunk c.
    if (c + 1 < NCHUNK) {
      tdm_load_2d(gSig + (c + 1) * 32, lds_addr_of(&S.work[(c + 1) & 1][0]), 32, 16, KTOT);
      tdm_load_2d(gRgb + (c + 1) * 96, lds_addr_of(&S.rgb [(c + 1) & 1][0]), 96, 16, KTOT * 3);
      __builtin_amdgcn_s_wait_tensorcnt(2);   // chunk c's pair has landed
    } else {
      __builtin_amdgcn_s_wait_tensorcnt(0);
    }
    __syncthreads();
#else
    __syncthreads();               // previous consumers of this buffer are done
    // ---- manual coalesced staging: sigma 16x32 f32, rgb 16x96 f32 --------
#pragma unroll
    for (int it = 0; it < 4; ++it) {
      int idx = L + 32 * it;                       // 0..127 float4 tiles
      int row = idx >> 3, col = idx & 7;
      float4 vv = *(const float4*)(gSig + row * KTOT + c * 32 + col * 4);
      *(float4*)(&WK[row * P_SIG + col * 4]) = vv;
    }
#pragma unroll
    for (int it = 0; it < 12; ++it) {
      int idx = L + 32 * it;                       // 0..383 float4 tiles
      int row = idx / 24, col = idx - row * 24;
      float4 vv = *(const float4*)(gRgb + row * (KTOT * 3) + c * 96 + col * 4);
      *(float4*)(&RB[row * P_RGB + col * 4 + ((col >> 3) << 2)]) = vv;
    }
    if (c + 1 < NCHUNK) {          // hint next chunk into cache (global_prefetch_b8)
      int prow = L >> 1, ph = L & 1;
      __builtin_prefetch(gSig + prow * KTOT + (c + 1) * 32 + ph * 16, 0, 1);
      __builtin_prefetch(gRgb + prow * (KTOT * 3) + (c + 1) * 96 + ph * 48, 0, 1);
    }
    __syncthreads();
#endif

    // ---- A-phase: alpha terms + log terms in WMMA-A layout ---------------
    v16h A;
    float ps0 = 0.0f, ps1 = 0.0f;
#pragma unroll
    for (int e = 0; e < 16; ++e) {
      int k  = e + 8 * h + ((e >= 8) ? 8 : 0);     // chunk-local sample
      int sg = c * 32 + k;                         // global sample
      float sig = WK[r * P_SIG + k];
      float delta;
      if (COARSE) delta = (sg == KTOT - 1) ? 1e10f : S.dL[r];
      else        delta = (sg == KTOT - 1) ? 1e10f : (S.z[r * P_Z + sg + 1] - S.z[r * P_Z + sg]);
      float y  = delta * fmaxf(sig, 0.0f);
      float ev = __expf(-y);                       // 1 - alpha
      WK[r * P_SIG + k] = ev;                      // same-lane in-place overwrite
      float lg = __logf(ev + 1e-10f);              // log(1-alpha+1e-10)
      A[e] = (_Float16)lg;
      if (e < 8) ps0 += lg; else ps1 += lg;
    }
    // per-ray 16-sample block totals (f32 carries)
    float bs0 = ps0 + __shfl_xor(ps0, 16, 32);     // sum of Lg over K=0..15
    float bs1 = ps1 + __shfl_xor(ps1, 16, 32);     // sum of Lg over K=16..31

    // ---- WMMA: exclusive prefix sums within the 32-sample chunk ----------
    v8f d0 = {}; v8f d1 = {};
    d0 = __builtin_amdgcn_wmma_f32_16x16x32_f16(false, A, false, B0, (short)0, d0, false, false);
    d1 = __builtin_amdgcn_wmma_f32_16x16x32_f16(false, A, false, B1, (short)0, d1, false, false);

    __syncthreads();                               // e-values visible cross-lane

    // ---- D-phase: weights + streaming reductions (D layout) --------------
#pragma unroll
    for (int ob = 0; ob < 2; ++ob) {
      float carry_b = carry + (ob ? bs0 : 0.0f);
      int sl = (L & 15) + 16 * ob;                 // chunk-local sample = column N
      int sg = c * 32 + sl;
#pragma unroll
      for (int v = 0; v < 8; ++v) {
        int rv = v + 8 * h;                        // matrix row M = ray
        float cv = __shfl(carry_b, rv, 32);        // carry of ray rv
        float tl = cv + (ob ? d1[v] : d0[v]);      // log T_exclusive
        float ev = WK[rv * P_SIG + sl];
        float al = 1.0f - ev;
        float w  = al * __expf(tl);
        if (COARSE) S.w[rv * P_W + sg] = w;        // needed by importance sampling
        acc_w[v] += w;
        acc_r[v] = fmaf(w, RB[rgb_idx(rv, sl * 3 + 0)], acc_r[v]);
        acc_g[v] = fmaf(w, RB[rgb_idx(rv, sl * 3 + 1)], acc_g[v]);
        acc_b[v] = fmaf(w, RB[rgb_idx(rv, sl * 3 + 2)], acc_b[v]);
        float z = COARSE ? (S.nearL[rv] + S.dL[rv] * (float)sg) : S.z[rv * P_Z + sg];
        acc_z[v] = fmaf(w, z, acc_z[v]);
      }
    }
    carry += bs0 + bs1;
  }

  // ---- reduce over the 16 sample-columns held across each lane half ------
#pragma unroll
  for (int m = 1; m < 16; m <<= 1) {
#pragma unroll
    for (int v = 0; v < 8; ++v) {
      acc_w[v] += __shfl_xor(acc_w[v], m, 32);
      acc_r[v] += __shfl_xor(acc_r[v], m, 32);
      acc_g[v] += __shfl_xor(acc_g[v], m, 32);
      acc_b[v] += __shfl_xor(acc_b[v], m, 32);
      acc_z[v] += __shfl_xor(acc_z[v], m, 32);
    }
  }

  // ---- finalize: white background, depth fill-to-far ---------------------
  const size_t N = (size_t)nRays;
  float* oR = out + (COARSE ? (size_t)0 : 5 * N);
  float* oA = out + (COARSE ? 3 * N : 8 * N);
  float* oD = out + (COARSE ? 4 * N : 9 * N);
#pragma unroll
  for (int v = 0; v < 8; ++v) {
    if ((L & 15) == v) {
      int rv   = v + 8 * h;
      int gray = ray0 + rv;
      float ws = acc_w[v];
      float bg = 1.0f - ws;
      oR[(size_t)gray * 3 + 0] = acc_r[v] + bg;
      oR[(size_t)gray * 3 + 1] = acc_g[v] + bg;
      oR[(size_t)gray * 3 + 2] = acc_b[v] + bg;
      oA[gray] = ws;
      oD[gray] = acc_z[v] + bg * S.farL[rv];
    }
  }
}

// ---------------------------------------------------------------------------
__global__ void __launch_bounds__(32)
nerf_render_kernel(const float* __restrict__ rays,
                   const float* __restrict__ sigC, const float* __restrict__ rgbC,
                   const float* __restrict__ sigF, const float* __restrict__ rgbF,
                   float* __restrict__ out, int nRays)
{
  __shared__ SharedBufs S;
  const int L    = threadIdx.x;
  const int ray0 = blockIdx.x * 16;

  // Only near/far of the ray tuple influence any output.
  if (L < 16) {
    float nr = rays[(size_t)(ray0 + L) * 8 + 6];
    float fr = rays[(size_t)(ray0 + L) * 8 + 7];
    S.nearL[L] = nr; S.farL[L] = fr; S.dL[L] = (fr - nr) * (1.0f / 64.0f);
  }

  // Constant strictly-lower-triangular B matrices (scan operators), f16-exact.
  // B0: outputs s=0..15 (t < s); B1: outputs s=16..31 (t < s+16).
  v16h B0, B1;
  {
    int n  = L & 15;
    int kb = (L >> 4) * 8;
#pragma unroll
    for (int i = 0; i < 16; ++i) {
      int t = kb + i + ((i >= 8) ? 8 : 0);
      B0[i] = (t < n)      ? (_Float16)1.0f : (_Float16)0.0f;
      B1[i] = (t < n + 16) ? (_Float16)1.0f : (_Float16)0.0f;
    }
  }
  __syncthreads();

  // -------- pass 1: coarse composite (uniform z, delta = (far-near)/64) ----
  composite_pass<NCOARSE, true>(L, sigC + (size_t)ray0 * NCOARSE,
                                rgbC + (size_t)ray0 * NCOARSE * 3,
                                S, B0, B1, out, ray0, nRays);
  __syncthreads();

  // -------- importance sampling: unnormalized CDF over w[1..62] ------------
  float* sCdf = &S.rgb[0][0];                // reuse rgb staging as CDF [16][P_W]
  if (L < 16) {
    float acc = 0.0f;
    sCdf[L * P_W + 0] = 0.0f;
#pragma unroll 1
    for (int i = 1; i <= 62; ++i) {
      acc += S.w[L * P_W + i] + 1e-5f;       // pdf_j = (w[j+1]+EPS)/S
      sCdf[L * P_W + i] = acc;
    }
    S.SL[L] = acc;
    S.invSL[L] = 1.0f / acc;
  }
  __syncthreads();

  // -------- z_fine via inverse CDF (det, u = i/63), written into S.w -------
  float* sZf = S.w;
#pragma unroll 1
  for (int it = 0; it < 32; ++it) {
    int task = L + 32 * it;                  // 16 rays x 64 u-values
    int rr = task >> 6, i = task & 63;
    float u   = (float)i * (1.0f / 63.0f);
    float tgt = u * S.SL[rr];
    int lo = 0, hi = 63;                     // upper_bound over 63-entry CDF
    while (lo < hi) { int mid = (lo + hi) >> 1; if (sCdf[rr * P_W + mid] <= tgt) lo = mid + 1; else hi = mid; }
    int below = (lo - 1 > 0) ? lo - 1 : 0;
    int above = (lo < 62) ? lo : 62;
    float iS = S.invSL[rr];
    float cb = sCdf[rr * P_W + below] * iS;
    float ca = sCdf[rr * P_W + above] * iS;
    float nr = S.nearL[rr], dd = S.dL[rr];
    float bb = nr + dd * ((float)below + 0.5f);   // bins = z midpoints
    float ba = nr + dd * ((float)above + 0.5f);
    float den = ca - cb;
    den = (den < 1e-5f) ? 1.0f : den;
    sZf[rr * P_W + i] = bb + (u - cb) / den * (ba - bb);
  }
  __syncthreads();

  // -------- stable merge of z_coarse (analytic) and z_fine into S.z --------
  // rank = own index + count of other list before it; identical float
  // comparators on both sides guarantee a bijection of positions.
#pragma unroll 1
  for (int it = 0; it < 32; ++it) {          // fine elements
    int task = L + 32 * it;
    int rr = task >> 6, j = task & 63;
    float zf = sZf[rr * P_W + j];
    float nr = S.nearL[rr], dd = S.dL[rr];
    int lo = 0, hi = 64;                     // count coarse (nr+dd*i) <= zf
    while (lo < hi) { int mid = (lo + hi) >> 1; if (nr + dd * (float)mid <= zf) lo = mid + 1; else hi = mid; }
    S.z[rr * P_Z + j + lo] = zf;
  }
#pragma unroll 1
  for (int it = 0; it < 32; ++it) {          // coarse elements
    int task = L + 32 * it;
    int rr = task >> 6, i = task & 63;
    float nr = S.nearL[rr], dd = S.dL[rr];
    float zc = nr + dd * (float)i;
    int lo = 0, hi = 64;                     // count fine < zc (strict)
    while (lo < hi) { int mid = (lo + hi) >> 1; if (sZf[rr * P_W + mid] < zc) lo = mid + 1; else hi = mid; }
    S.z[rr * P_Z + i + lo] = zc;
  }
  __syncthreads();

  // -------- pass 2: fine composite over merged 128 samples ------------------
  composite_pass<NTOTAL, false>(L, sigF + (size_t)ray0 * NTOTAL,
                                rgbF + (size_t)ray0 * NTOTAL * 3,
                                S, B0, B1, out, ray0, nRays);
}

// ---------------------------------------------------------------------------
extern "C" void kernel_launch(void* const* d_in, const int* in_sizes, int n_in,
                              void* d_out, int out_size, void* d_ws, size_t ws_size,
                              hipStream_t stream) {
  (void)n_in; (void)out_size; (void)d_ws; (void)ws_size;
  const float* rays = (const float*)d_in[0];
  const float* sigC = (const float*)d_in[1];
  const float* rgbC = (const float*)d_in[2];
  const float* sigF = (const float*)d_in[3];
  const float* rgbF = (const float*)d_in[4];
  float* out = (float*)d_out;

  int nRays = in_sizes[1] / NCOARSE;         // sigmas_coarse = BS*N_RAYS*64
  int nWaves = nRays / 16;                   // one wave32 per 16 rays

  nerf_render_kernel<<<dim3(nWaves), dim3(32), 0, stream>>>(
      rays, sigC, rgbC, sigF, rgbF, out, nRays);
}
